// ParallelSARDecoder_12421045420503
// MI455X (gfx1250) — compile-verified, matrix operand order checked by reading, not run on hardware
//
#include <hip/hip_runtime.h>
#include <hip/hip_bf16.h>

// ---------------------------------------------------------------------------
// MI455X (gfx1250) ParallelSAR decoder forward pass.
// All dense contractions run on v_wmma_f32_16x16x32_bf16 (bf16 in, f32 acc).
// This revision register-blocks the GEMMs 2x2 (32x32 per wave: 4 WMMA per
// 8 b128 loads instead of 1 per 4) and fixes prefetch locality to WGP scope.
// ---------------------------------------------------------------------------

typedef __bf16 bf16;
typedef bf16  bfx16 __attribute__((ext_vector_type(16)));
typedef bf16  bfx8  __attribute__((ext_vector_type(8)));
typedef float fx8   __attribute__((ext_vector_type(8)));

#define B_   64
#define T_   30
#define T1_  31
#define DM_  512
#define DK_  64
#define H_   8
#define W_   64
#define HW_  512
#define G_   2048          // 4*DM
#define KC_  4608          // 9*DM (conv implicit-GEMM K)
#define PR_  92            // NC-1

union UV { bfx16 v; bfx8 h[2]; };

// ---------------------------------------------------------------------------
// Generic WMMA GEMM:  C[m,n] = sum_k A[m,k]*Wt[n,k]  (+ Cinit[m,n] + bias[n])
// A, Wt bf16 row-major.  One wave computes a 32x32 macro-tile as 2x2 WMMA
// tiles (4 v_wmma per K-step, A/B fragments each reused twice).
// Operand layout (ISA 7.12.2, 16-bit A 16x32): lane L holds row (L&15);
// K chunks [hf*8, hf*8+8) and [16+hf*8, 16+hf*8+8) where hf = L>>4.
// C/D layout: lane L holds column (L&15); VGPR r holds M = r + 8*(L>>4).
// ---------------------------------------------------------------------------
__global__ void wmma_gemm_bf16(const bf16* __restrict__ A,
                               const bf16* __restrict__ Wt,
                               const float* __restrict__ Cinit,
                               const float* __restrict__ bias,
                               float* __restrict__ C,
                               bf16* __restrict__ Cb,
                               int M, int N, int K,
                               int lda, int ldw, int ldci, int ldc,
                               long sA, long sW, long sCi, long sC) {
  const int lane = threadIdx.x & 31;
  const int wave = threadIdx.x >> 5;
  const int tilesN = (N + 31) >> 5;                  // 32-wide macro tiles
  const int tilesM = (M + 31) >> 5;
  const int tile = blockIdx.x * 4 + wave;
  if (tile >= tilesM * tilesN) return;               // wave-uniform exit
  const int tm = tile / tilesN;
  const int tn = tile % tilesN;

  const long z = blockIdx.y;
  A += z * sA;  Wt += z * sW;
  const float* Ci = Cinit ? Cinit + z * sCi : nullptr;
  float* Cp = C ? C + z * sC : nullptr;
  bf16*  Cq = Cb ? Cb + z * sC : nullptr;

  const int r  = lane & 15;
  const int hf = lane >> 4;
  // clamped row bases for edge tiles (loads stay in-bounds; stores guarded)
  int ra0 = tm * 32 + r;       if (ra0 > M - 1) ra0 = M - 1;
  int ra1 = tm * 32 + 16 + r;  if (ra1 > M - 1) ra1 = M - 1;
  int rb0 = tn * 32 + r;       if (rb0 > N - 1) rb0 = N - 1;
  int rb1 = tn * 32 + 16 + r;  if (rb1 > N - 1) rb1 = N - 1;
  const bf16* ap0 = A  + (long)ra0 * lda;
  const bf16* ap1 = A  + (long)ra1 * lda;
  const bf16* bp0 = Wt + (long)rb0 * ldw;
  const bf16* bp1 = Wt + (long)rb1 * ldw;
  const int n0 = tn * 32 + r;                        // lane's columns
  const int n1 = n0 + 16;

  auto initv = [&](int m, int n) -> float {
    float v = 0.f;
    if (m < M && n < N) {
      if (Ci)   v  = Ci[(long)m * ldci + n];
      if (bias) v += bias[n];
    }
    return v;
  };
  fx8 acc00, acc01, acc10, acc11;
#pragma unroll
  for (int i = 0; i < 8; ++i) {
    const int m0 = tm * 32 + i + 8 * hf;
    const int m1 = m0 + 16;
    acc00[i] = initv(m0, n0);
    acc01[i] = initv(m0, n1);
    acc10[i] = initv(m1, n0);
    acc11[i] = initv(m1, n1);
  }

  for (int k0 = 0; k0 < K; k0 += 32) {
    UV a0, a1, b0, b1;
    a0.h[0] = *reinterpret_cast<const bfx8*>(ap0 + k0 + hf * 8);
    a0.h[1] = *reinterpret_cast<const bfx8*>(ap0 + k0 + 16 + hf * 8);
    a1.h[0] = *reinterpret_cast<const bfx8*>(ap1 + k0 + hf * 8);
    a1.h[1] = *reinterpret_cast<const bfx8*>(ap1 + k0 + 16 + hf * 8);
    b0.h[0] = *reinterpret_cast<const bfx8*>(bp0 + k0 + hf * 8);
    b0.h[1] = *reinterpret_cast<const bfx8*>(bp0 + k0 + 16 + hf * 8);
    b1.h[0] = *reinterpret_cast<const bfx8*>(bp1 + k0 + hf * 8);
    b1.h[1] = *reinterpret_cast<const bfx8*>(bp1 + k0 + 16 + hf * 8);
    __builtin_prefetch(bp0 + k0 + 256, 0, 3);        // WGP-scope prefetch
    __builtin_prefetch(bp1 + k0 + 256, 0, 3);
    acc00 = __builtin_amdgcn_wmma_f32_16x16x32_bf16(false, a0.v, false, b0.v,
                                                    (short)0, acc00, false, false);
    acc01 = __builtin_amdgcn_wmma_f32_16x16x32_bf16(false, a0.v, false, b1.v,
                                                    (short)0, acc01, false, false);
    acc10 = __builtin_amdgcn_wmma_f32_16x16x32_bf16(false, a1.v, false, b0.v,
                                                    (short)0, acc10, false, false);
    acc11 = __builtin_amdgcn_wmma_f32_16x16x32_bf16(false, a1.v, false, b1.v,
                                                    (short)0, acc11, false, false);
  }

  auto storev = [&](int m, int n, float v) {
    if (m < M && n < N) {
      const long idx = (long)m * ldc + n;
      if (Cp) Cp[idx] = v;
      if (Cq) Cq[idx] = (bf16)v;
    }
  };
#pragma unroll
  for (int i = 0; i < 8; ++i) {
    const int m0 = tm * 32 + i + 8 * hf;
    const int m1 = m0 + 16;
    storev(m0, n0, acc00[i]);
    storev(m0, n1, acc01[i]);
    storev(m1, n0, acc10[i]);
    storev(m1, n1, acc11[i]);
  }
}

// ---------------------------------------------------------------------------
// Implicit-GEMM 3x3 SAME conv: featN [B][H][W][C] bf16, Wr [DK][9*C] bf16.
// K ordered (dy,dx) outer (9 blocks of C=512, so each 32-wide K window stays
// inside one tap).  One wave owns 16 pixels x all 64 outputs: the single A
// fragment is reused across 4 B tiles -> 4 WMMA per 10 b128 loads.
// Output Kout [B][H][W][DK] f32 feeds the attention-score kernel directly.
// ---------------------------------------------------------------------------
__global__ void wmma_conv3x3(const bf16* __restrict__ featN,
                             const bf16* __restrict__ Wr,
                             const float* __restrict__ bias,
                             float* __restrict__ Kout) {
  const int lane = threadIdx.x & 31;
  const int wave = threadIdx.x >> 5;
  const int tm = blockIdx.x * 4 + wave;              // pixel tile

  const int r  = lane & 15;
  const int hf = lane >> 4;
  const int p = tm * 16 + r;                         // pixel id (M mult of 16)
  const int b = p / (H_ * W_);
  const int hw = p % (H_ * W_);
  const int h = hw / W_, w = hw % W_;
  const bf16* bp0 = Wr + (long)(r) * KC_;            // B rows r, r+16, r+32, r+48
  const bf16* bp1 = Wr + (long)(16 + r) * KC_;
  const bf16* bp2 = Wr + (long)(32 + r) * KC_;
  const bf16* bp3 = Wr + (long)(48 + r) * KC_;

  fx8 acc0, acc1, acc2, acc3;
#pragma unroll
  for (int i = 0; i < 8; ++i) {
    acc0[i] = bias[r];
    acc1[i] = bias[16 + r];
    acc2[i] = bias[32 + r];
    acc3[i] = bias[48 + r];
  }

  for (int k0 = 0; k0 < KC_; k0 += 32) {
    const int blk = k0 >> 9;                         // tap index 0..8
    const int dy = blk / 3 - 1, dx = blk % 3 - 1;
    const int cc = k0 & (DM_ - 1);
    UV a, b0, b1, b2, b3;
#pragma unroll
    for (int i = 0; i < 16; ++i) a.v[i] = (bf16)0.0f;
    const int ys = h + dy, xs = w + dx;
    if ((unsigned)ys < (unsigned)H_ && (unsigned)xs < (unsigned)W_) {
      const bf16* src = featN + (((long)(b * H_ + ys)) * W_ + xs) * DM_ + cc;
      a.h[0] = *reinterpret_cast<const bfx8*>(src + hf * 8);
      a.h[1] = *reinterpret_cast<const bfx8*>(src + 16 + hf * 8);
    }
    b0.h[0] = *reinterpret_cast<const bfx8*>(bp0 + k0 + hf * 8);
    b0.h[1] = *reinterpret_cast<const bfx8*>(bp0 + k0 + 16 + hf * 8);
    b1.h[0] = *reinterpret_cast<const bfx8*>(bp1 + k0 + hf * 8);
    b1.h[1] = *reinterpret_cast<const bfx8*>(bp1 + k0 + 16 + hf * 8);
    b2.h[0] = *reinterpret_cast<const bfx8*>(bp2 + k0 + hf * 8);
    b2.h[1] = *reinterpret_cast<const bfx8*>(bp2 + k0 + 16 + hf * 8);
    b3.h[0] = *reinterpret_cast<const bfx8*>(bp3 + k0 + hf * 8);
    b3.h[1] = *reinterpret_cast<const bfx8*>(bp3 + k0 + 16 + hf * 8);
    acc0 = __builtin_amdgcn_wmma_f32_16x16x32_bf16(false, a.v, false, b0.v,
                                                   (short)0, acc0, false, false);
    acc1 = __builtin_amdgcn_wmma_f32_16x16x32_bf16(false, a.v, false, b1.v,
                                                   (short)0, acc1, false, false);
    acc2 = __builtin_amdgcn_wmma_f32_16x16x32_bf16(false, a.v, false, b2.v,
                                                   (short)0, acc2, false, false);
    acc3 = __builtin_amdgcn_wmma_f32_16x16x32_bf16(false, a.v, false, b3.v,
                                                   (short)0, acc3, false, false);
  }

#pragma unroll
  for (int i = 0; i < 8; ++i) {
    const int m = tm * 16 + i + 8 * hf;              // pixel row
    float* o = Kout + (long)m * DK_;
    o[r]      = acc0[i];
    o[16 + r] = acc1[i];
    o[32 + r] = acc2[i];
    o[48 + r] = acc3[i];
  }
}

// ------------------------- small helper kernels ----------------------------
__global__ void cast_f32_bf16(bf16* __restrict__ d, const float* __restrict__ s, int n) {
  int i = blockIdx.x * blockDim.x + threadIdx.x;
  if (i < n) d[i] = (bf16)s[i];
}
__global__ void zero_f32(float* p, int n) {
  int i = blockIdx.x * blockDim.x + threadIdx.x;
  if (i < n) p[i] = 0.f;
}
__global__ void zero_bf16(bf16* p, int n) {
  int i = blockIdx.x * blockDim.x + threadIdx.x;
  if (i < n) p[i] = (bf16)0.0f;
}
__global__ void add_bias_vec(float* d, const float* a, const float* b, int n) {
  int i = blockIdx.x * blockDim.x + threadIdx.x;
  if (i < n) d[i] = a[i] + b[i];
}
// conv_w [DK][DM][3][3] f32 -> [DK][9][DM] bf16 (tap-major K)
__global__ void rearrange_conv_w(bf16* __restrict__ d, const float* __restrict__ s) {
  int i = blockIdx.x * blockDim.x + threadIdx.x;
  if (i >= DK_ * KC_) return;
  const int n = i / KC_, kk = i % KC_;
  const int tap = kk / DM_, c = kk % DM_;
  d[i] = (bf16)s[((long)(n * DM_ + c)) * 9 + tap];
}
// feat [B][C][H][W] f32 -> [B][H][W][C] bf16
__global__ void feat_to_nhwc(bf16* __restrict__ d, const float* __restrict__ s) {
  int i = blockIdx.x * blockDim.x + threadIdx.x;
  if (i >= B_ * H_ * W_ * DM_) return;
  const int c = i % DM_;
  const int w = (i / DM_) % W_;
  const int h = (i / (DM_ * W_)) % H_;
  const int b = i / (DM_ * W_ * H_);
  d[i] = (bf16)s[(((long)(b * DM_ + c)) * H_ + h) * W_ + w];
}
// dec_in bf16 [B][T1][DM]: t==0 -> out_enc, else emb[label[b][t-1]]
__global__ void build_decin(bf16* __restrict__ d, const float* __restrict__ oe,
                            const float* __restrict__ emb, const int* __restrict__ lab) {
  int i = blockIdx.x * blockDim.x + threadIdx.x;
  if (i >= B_ * T1_ * DM_) return;
  const int c = i % DM_;
  const int t = (i / DM_) % T1_;
  const int b = i / (DM_ * T1_);
  float v = (t == 0) ? oe[b * DM_ + c] : emb[(long)lab[b * T_ + (t - 1)] * DM_ + c];
  d[i] = (bf16)v;
}
// PyTorch-order LSTM cell: gates [B][4*DM] (i,f,g,o); updates c, emits h(bf16)
__global__ void lstm_cell(const float* __restrict__ gates, float* __restrict__ c,
                          bf16* __restrict__ hb, bf16* __restrict__ yb, int t) {
  int i = blockIdx.x * blockDim.x + threadIdx.x;
  if (i >= B_ * DM_) return;
  const int b = i / DM_, d = i % DM_;
  const float* g = gates + (long)b * G_;
  const float gi = 1.f / (1.f + __expf(-g[d]));
  const float gf = 1.f / (1.f + __expf(-g[DM_ + d]));
  const float gg = tanhf(g[2 * DM_ + d]);
  const float go = 1.f / (1.f + __expf(-g[3 * DM_ + d]));
  const float cn = gf * c[i] + gi * gg;
  const float hn = go * tanhf(cn);
  c[i] = cn;
  hb[i] = (bf16)hn;
  yb[((long)b * T1_ + t) * DM_ + d] = (bf16)hn;
}
// score = sum_c tanh(k[b,hw,c] + q[b,t,c]) * w2[c] + b2, width-masked softmax
__global__ void attn_softmax(const float* __restrict__ kf, const float* __restrict__ q,
                             const float* __restrict__ w2, const float* __restrict__ b2,
                             const float* __restrict__ ratios, bf16* __restrict__ attn) {
  __shared__ float sq[DK_], sw[DK_], red[256];
  const int b = blockIdx.x / T1_, t = blockIdx.x % T1_;
  const int tid = threadIdx.x;
  if (tid < DK_) {
    sq[tid] = q[((long)b * T1_ + t) * DK_ + tid];
    sw[tid] = w2[tid];
  }
  __syncthreads();
  int vw = (int)ceilf(W_ * ratios[b]); if (vw > W_) vw = W_;
  const float bb = b2[0];
  float s[2];
#pragma unroll
  for (int j = 0; j < 2; ++j) {
    const int pos = tid + j * 256;
    const float* kp = kf + ((long)b * HW_ + pos) * DK_;
    float acc = 0.f;
#pragma unroll 8
    for (int c = 0; c < DK_; ++c) acc += tanhf(kp[c] + sq[c]) * sw[c];
    acc += bb;
    if ((pos % W_) >= vw) acc = -3.0e38f;            // masked -> -inf
    s[j] = acc;
  }
  red[tid] = fmaxf(s[0], s[1]); __syncthreads();
  for (int off = 128; off > 0; off >>= 1) {
    if (tid < off) red[tid] = fmaxf(red[tid], red[tid + off]);
    __syncthreads();
  }
  const float mx = red[0]; __syncthreads();
  const float e0 = (s[0] <= -3.0e38f) ? 0.f : __expf(s[0] - mx);
  const float e1 = (s[1] <= -3.0e38f) ? 0.f : __expf(s[1] - mx);
  red[tid] = e0 + e1; __syncthreads();
  for (int off = 128; off > 0; off >>= 1) {
    if (tid < off) red[tid] += red[tid + off];
    __syncthreads();
  }
  const float inv = 1.f / red[0];
  bf16* o = attn + ((long)b * T1_ + t) * HW_;
  o[tid] = (bf16)(e0 * inv);
  o[tid + 256] = (bf16)(e1 * inv);
}
// pred_in bf16 [B*T1][1536] = concat(y1, attn_feat, holistic=out_enc)
__global__ void concat_predin(bf16* __restrict__ d, const bf16* __restrict__ y,
                              const bf16* __restrict__ af, const float* __restrict__ oe) {
  int i = blockIdx.x * blockDim.x + threadIdx.x;
  if (i >= B_ * T1_ * 3 * DM_) return;
  const int row = i / (3 * DM_), col = i % (3 * DM_);
  const int b = row / T1_;
  bf16 v;
  if (col < DM_)            v = y[(long)row * DM_ + col];
  else if (col < 2 * DM_)   v = af[(long)row * DM_ + col - DM_];
  else                      v = (bf16)oe[b * DM_ + col - 2 * DM_];
  d[i] = v;
}
// out [B][T][92] = pred[:, 1:, :]
__global__ void final_copy(float* __restrict__ out, const float* __restrict__ pred) {
  int i = blockIdx.x * blockDim.x + threadIdx.x;
  if (i >= B_ * T_ * PR_) return;
  const int n = i % PR_;
  const int t = (i / PR_) % T_;
  const int b = i / (PR_ * T_);
  out[i] = pred[((long)b * T1_ + (t + 1)) * PR_ + n];
}

// ---------------------------------------------------------------------------
static inline void launch_gemm(hipStream_t s, const bf16* A, const bf16* Wt,
                               const float* Ci, const float* bias,
                               float* C, bf16* Cb,
                               int M, int N, int K, int lda, int ldw, int ldci,
                               int ldc, long sA, long sW, long sCi, long sC,
                               int batch) {
  const int tiles = ((M + 31) / 32) * ((N + 31) / 32);
  dim3 grid((tiles + 3) / 4, batch);
  wmma_gemm_bf16<<<grid, 128, 0, s>>>(A, Wt, Ci, bias, C, Cb, M, N, K,
                                      lda, ldw, ldci, ldc, sA, sW, sCi, sC);
}

extern "C" void kernel_launch(void* const* d_in, const int* in_sizes, int n_in,
                              void* d_out, int out_size, void* d_ws, size_t ws_size,
                              hipStream_t stream) {
  const float* feat   = (const float*)d_in[0];
  const float* oe     = (const float*)d_in[1];
  const int*   label  = (const int*)d_in[2];
  const float* ratios = (const float*)d_in[3];
  const float* emb    = (const float*)d_in[4];
  const float* w_ih0  = (const float*)d_in[5];
  const float* w_hh0  = (const float*)d_in[6];
  const float* b_ih0  = (const float*)d_in[7];
  const float* b_hh0  = (const float*)d_in[8];
  const float* w_ih1  = (const float*)d_in[9];
  const float* w_hh1  = (const float*)d_in[10];
  const float* b_ih1  = (const float*)d_in[11];
  const float* b_hh1  = (const float*)d_in[12];
  const float* wq     = (const float*)d_in[13];
  const float* bq     = (const float*)d_in[14];
  const float* conv_w = (const float*)d_in[15];
  const float* conv_b = (const float*)d_in[16];
  const float* w2     = (const float*)d_in[17];
  const float* b2     = (const float*)d_in[18];
  const float* wp     = (const float*)d_in[19];
  const float* bp     = (const float*)d_in[20];
  float* out = (float*)d_out;

  char* ws = (char*)d_ws;
  size_t off = 0;
  auto alloc = [&](size_t bytes) -> void* {
    void* p = ws + off;
    off = (off + bytes + 255) & ~(size_t)255;
    return p;
  };
  const int NWIH = G_ * DM_;                 // 2048*512
  bf16* wih0b = (bf16*)alloc((size_t)NWIH * 2);
  bf16* whh0b = (bf16*)alloc((size_t)NWIH * 2);
  bf16* wih1b = (bf16*)alloc((size_t)NWIH * 2);
  bf16* whh1b = (bf16*)alloc((size_t)NWIH * 2);
  bf16* wqb   = (bf16*)alloc((size_t)DK_ * DM_ * 2);
  bf16* wpb   = (bf16*)alloc((size_t)PR_ * 3 * DM_ * 2);
  bf16* wcb   = (bf16*)alloc((size_t)DK_ * KC_ * 2);
  bf16* featN = (bf16*)alloc((size_t)B_ * H_ * W_ * DM_ * 2);
  bf16* featC = (bf16*)alloc((size_t)B_ * DM_ * HW_ * 2);
  bf16* decb  = (bf16*)alloc((size_t)B_ * T1_ * DM_ * 2);
  bf16* y0b   = (bf16*)alloc((size_t)B_ * T1_ * DM_ * 2);
  bf16* y1b   = (bf16*)alloc((size_t)B_ * T1_ * DM_ * 2);
  bf16* afb   = (bf16*)alloc((size_t)B_ * T1_ * DM_ * 2);
  bf16* attnb = (bf16*)alloc((size_t)B_ * T1_ * HW_ * 2);
  bf16* pinb  = (bf16*)alloc((size_t)B_ * T1_ * 3 * DM_ * 2);
  bf16* hb    = (bf16*)alloc((size_t)B_ * DM_ * 2);
  float* xg   = (float*)alloc((size_t)B_ * T1_ * G_ * 4);
  float* gates= (float*)alloc((size_t)B_ * G_ * 4);
  float* cbuf = (float*)alloc((size_t)B_ * DM_ * 4);
  float* qf   = (float*)alloc((size_t)B_ * T1_ * DK_ * 4);
  float* kf   = (float*)alloc((size_t)B_ * HW_ * DK_ * 4);
  float* pred = (float*)alloc((size_t)B_ * T1_ * PR_ * 4);
  float* bias0= (float*)alloc((size_t)G_ * 4);
  float* bias1= (float*)alloc((size_t)G_ * 4);

  auto blk = [](int n) { return (n + 255) / 256; };

  // ---- weight/feature conversion ----
  cast_f32_bf16<<<blk(NWIH), 256, 0, stream>>>(wih0b, w_ih0, NWIH);
  cast_f32_bf16<<<blk(NWIH), 256, 0, stream>>>(whh0b, w_hh0, NWIH);
  cast_f32_bf16<<<blk(NWIH), 256, 0, stream>>>(wih1b, w_ih1, NWIH);
  cast_f32_bf16<<<blk(NWIH), 256, 0, stream>>>(whh1b, w_hh1, NWIH);
  cast_f32_bf16<<<blk(DK_ * DM_), 256, 0, stream>>>(wqb, wq, DK_ * DM_);
  cast_f32_bf16<<<blk(PR_ * 3 * DM_), 256, 0, stream>>>(wpb, wp, PR_ * 3 * DM_);
  rearrange_conv_w<<<blk(DK_ * KC_), 256, 0, stream>>>(wcb, conv_w);
  const int NF = B_ * H_ * W_ * DM_;
  feat_to_nhwc<<<blk(NF), 256, 0, stream>>>(featN, feat);
  cast_f32_bf16<<<blk(NF), 256, 0, stream>>>(featC, feat, NF);   // BCHW copy
  build_decin<<<blk(B_ * T1_ * DM_), 256, 0, stream>>>(decb, oe, emb, label);
  add_bias_vec<<<blk(G_), 256, 0, stream>>>(bias0, b_ih0, b_hh0, G_);
  add_bias_vec<<<blk(G_), 256, 0, stream>>>(bias1, b_ih1, b_hh1, G_);

  const int M = B_ * T1_;                      // 1984

  // ---- LSTM layer 0 ----
  launch_gemm(stream, decb, wih0b, nullptr, bias0, xg, nullptr,
              M, G_, DM_, DM_, DM_, 0, G_, 0, 0, 0, 0, 1);
  zero_f32<<<blk(B_ * DM_), 256, 0, stream>>>(cbuf, B_ * DM_);
  zero_bf16<<<blk(B_ * DM_), 256, 0, stream>>>(hb, B_ * DM_);
  for (int t = 0; t < T1_; ++t) {
    launch_gemm(stream, hb, whh0b, xg + (long)t * G_, nullptr, gates, nullptr,
                B_, G_, DM_, DM_, DM_, T1_ * G_, G_, 0, 0, 0, 0, 1);
    lstm_cell<<<blk(B_ * DM_), 256, 0, stream>>>(gates, cbuf, hb, y0b, t);
  }

  // ---- LSTM layer 1 ----
  launch_gemm(stream, y0b, wih1b, nullptr, bias1, xg, nullptr,
              M, G_, DM_, DM_, DM_, 0, G_, 0, 0, 0, 0, 1);
  zero_f32<<<blk(B_ * DM_), 256, 0, stream>>>(cbuf, B_ * DM_);
  zero_bf16<<<blk(B_ * DM_), 256, 0, stream>>>(hb, B_ * DM_);
  for (int t = 0; t < T1_; ++t) {
    launch_gemm(stream, hb, whh1b, xg + (long)t * G_, nullptr, gates, nullptr,
                B_, G_, DM_, DM_, DM_, T1_ * G_, G_, 0, 0, 0, 0, 1);
    lstm_cell<<<blk(B_ * DM_), 256, 0, stream>>>(gates, cbuf, hb, y1b, t);
  }

  // ---- q = y @ wq^T + bq ----
  launch_gemm(stream, y1b, wqb, nullptr, bq, qf, nullptr,
              M, DK_, DM_, DM_, DM_, 0, DK_, 0, 0, 0, 0, 1);

  // ---- 3x3 conv as implicit WMMA GEMM (one wave = 16 pixels x 64 outs) ----
  {
    const int ptiles = B_ * H_ * W_ / 16;            // 2048 pixel tiles
    wmma_conv3x3<<<ptiles / 4, 128, 0, stream>>>(featN, wcb, conv_b, kf);
  }

  // ---- attention score + masked softmax ----
  attn_softmax<<<B_ * T1_, 256, 0, stream>>>(kf, qf, w2, b2, ratios, attnb);

  // ---- attn_feat[b] = attn[b] (T1 x HW) @ feat[b]^T (HW x DM), batched ----
  launch_gemm(stream, attnb, featC, nullptr, nullptr, nullptr, afb,
              T1_, DM_, HW_, HW_, HW_, 0, DM_,
              (long)T1_ * HW_, (long)DM_ * HW_, 0, (long)T1_ * DM_, B_);

  // ---- prediction head ----
  concat_predin<<<blk(B_ * T1_ * 3 * DM_), 256, 0, stream>>>(pinb, y1b, afb, oe);
  launch_gemm(stream, pinb, wpb, nullptr, bp, pred, nullptr,
              M, PR_, 3 * DM_, 3 * DM_, 3 * DM_, 0, PR_, 0, 0, 0, 0, 1);
  final_copy<<<blk(B_ * T_ * PR_), 256, 0, stream>>>(out, pred);
}
